// MultiViewReconstructionLoss_17205638988055
// MI455X (gfx1250) — compile-verified, matrix operand order checked by reading, not run on hardware
//
#include <hip/hip_runtime.h>
#include <hip/hip_bf16.h>

typedef __attribute__((ext_vector_type(16))) _Float16 v16h;
typedef __attribute__((ext_vector_type(8)))  float    v8f;

#define NN 4096
#define VV 4
#define CC 256
#define LDSTRIDE 40   // halfs per LDS row: 80B stride -> conflict-free b128 reads

union H16 { v16h v; float4 f4[2]; };

__device__ inline v16h load16h(const _Float16* p0, const _Float16* p1) {
    H16 u;
    u.f4[0] = *(const float4*)p0;
    u.f4[1] = *(const float4*)p1;
    return u.v;
}

__device__ inline v8f wmma_f16(v16h a, v16h b, v8f c) {
    return __builtin_amdgcn_wmma_f32_16x16x32_f16(false, a, false, b, (short)0, c, false, false);
}

// ---------------- prep kernels ----------------

// normalize desc rows -> f16  (one block per (n,v) row)
__global__ void prepNorm(const float* __restrict__ desc, _Float16* __restrict__ dnh) {
    __shared__ float red[256];
    __shared__ float scale_s;
    int row = blockIdx.x;            // N*V rows
    int t = threadIdx.x;             // channel
    float d = desc[(size_t)row * CC + t];
    red[t] = d * d;
    __syncthreads();
    for (int s = 128; s > 0; s >>= 1) { if (t < s) red[t] += red[t + s]; __syncthreads(); }
    if (t == 0) scale_s = 1.0f / fmaxf(sqrtf(red[0]), 1e-12f);
    __syncthreads();
    dnh[(size_t)row * CC + t] = (_Float16)(d * scale_s);
}

// sf -> f16 transposed: sfT[(v*C + c)*N + n]
__global__ void prepSfT(const float* __restrict__ sf, _Float16* __restrict__ sfT) {
    size_t id = (size_t)blockIdx.x * 256 + threadIdx.x;   // N*V*C elements
    int c = (int)(id & 255);
    size_t nv = id >> 8;
    int v = (int)(nv & 3);
    size_t n = nv >> 2;
    sfT[((size_t)(v * CC + c)) * NN + n] = (_Float16)sf[id];
}

// W_rec -> f16 transposed: WrecT[n*C + k] = W_rec[k*C + n]
__global__ void prepWrec(const float* __restrict__ W, _Float16* __restrict__ Wt) {
    int id = blockIdx.x * 256 + threadIdx.x;  // C*C
    int n = id & 255, k = id >> 8;
    Wt[n * CC + k] = (_Float16)W[k * CC + n];
}

// relWT[p*C + c] = concat(T[i],T[j]) @ W_T  ; also zero sqacc
__global__ void prepRel(const float* __restrict__ T, const float* __restrict__ W_T,
                        float* __restrict__ relWT, float* __restrict__ sqacc) {
    int p = blockIdx.x;          // 16 (i*4+j, diagonal unused but computed)
    int i = p >> 2, j = p & 3;
    int t = threadIdx.x;         // channel
    if (t == 0) sqacc[p] = 0.0f;
    float s = 0.0f;
    #pragma unroll
    for (int k = 0; k < 16; ++k) s += T[i * 16 + k] * W_T[k * CC + t];
    #pragma unroll
    for (int k = 0; k < 16; ++k) s += T[j * 16 + k] * W_T[(16 + k) * CC + t];
    relWT[p * CC + t] = s;
}

// ---------------- Pass A: sim row stats (max / argmax / sumexp) ----------------
// one wave per (pair, 16-row tile); WMMA 16x16x32 f16 over full 4096 columns.
// Lane-local online softmax stats (branchless) in hot loop; double-buffered B
// fragments; single cross-lane merge at the end.
__global__ void __launch_bounds__(256, 1)
passA(const _Float16* __restrict__ dnh,
      int* __restrict__ idxRow, float* __restrict__ mRow, float* __restrict__ lRow) {
    int wid  = (blockIdx.x * blockDim.x + threadIdx.x) >> 5;
    int lane = threadIdx.x & 31;
    int q = wid >> 8;            // 0..11
    int rowTile = wid & 255;
    if (q >= 12) return;
    int i = q / 3, rr = q % 3;
    int j = rr + (rr >= i ? 1 : 0);

    int hf = lane >> 4, lr = lane & 15;
    int koff = hf * 8;

    // preload A fragments: rows of dn_i (16 rows x 256 k)
    const _Float16* pa = dnh + ((size_t)((rowTile * 16 + lr) * VV + i)) * CC;
    v16h a[8];
    #pragma unroll
    for (int kc = 0; kc < 8; ++kc)
        a[kc] = load16h(pa + kc * 32 + koff, pa + kc * 32 + 16 + koff);

    const _Float16* dbase = dnh + (size_t)(lr * VV + j) * CC;

    // lane-local online stats over this lane's column subset
    float ml[8], ll[8]; int il[8];
    #pragma unroll
    for (int r = 0; r < 8; ++r) { ml[r] = -1e30f; ll[r] = 0.0f; il[r] = 0; }

    v16h b0[8], b1[8];
    {
        const _Float16* pb = dbase;      // ct = 0
        #pragma unroll
        for (int kc = 0; kc < 8; ++kc)
            b0[kc] = load16h(pb + kc * 32 + koff, pb + kc * 32 + 16 + koff);
    }

    for (int ct = 0; ct < 256; ct += 2) {
        // prefetch tile ct+1
        {
            const _Float16* pb = dbase + (size_t)(ct + 1) * 16 * VV * CC;
            #pragma unroll
            for (int kc = 0; kc < 8; ++kc)
                b1[kc] = load16h(pb + kc * 32 + koff, pb + kc * 32 + 16 + koff);
        }
        // compute tile ct with b0
        {
            v8f c = {};
            #pragma unroll
            for (int kc = 0; kc < 8; ++kc)
                c = wmma_f16(a[kc], b0[kc], c);
            int col = ct * 16 + lr;
            #pragma unroll
            for (int r = 0; r < 8; ++r) {
                float v = c[r];
                float mnew = fmaxf(ml[r], v);
                ll[r] = ll[r] * __expf(ml[r] - mnew) + __expf(v - mnew);
                il[r] = (v > ml[r]) ? col : il[r];   // strict > keeps first occurrence
                ml[r] = mnew;
            }
        }
        // prefetch tile ct+2
        if (ct + 2 < 256) {
            const _Float16* pb = dbase + (size_t)(ct + 2) * 16 * VV * CC;
            #pragma unroll
            for (int kc = 0; kc < 8; ++kc)
                b0[kc] = load16h(pb + kc * 32 + koff, pb + kc * 32 + 16 + koff);
        }
        // compute tile ct+1 with b1
        {
            v8f c = {};
            #pragma unroll
            for (int kc = 0; kc < 8; ++kc)
                c = wmma_f16(a[kc], b1[kc], c);
            int col = (ct + 1) * 16 + lr;
            #pragma unroll
            for (int r = 0; r < 8; ++r) {
                float v = c[r];
                float mnew = fmaxf(ml[r], v);
                ll[r] = ll[r] * __expf(ml[r] - mnew) + __expf(v - mnew);
                il[r] = (v > ml[r]) ? col : il[r];
                ml[r] = mnew;
            }
        }
    }

    // single cross-lane merge (16 lanes own interleaved columns of each row)
    #pragma unroll
    for (int r = 0; r < 8; ++r) {
        float m = ml[r], l = ll[r]; int idx = il[r];
        #pragma unroll
        for (int md = 1; md < 16; md <<= 1) {
            float om = __shfl_xor(m, md, 32);
            float ol = __shfl_xor(l, md, 32);
            int   oi = __shfl_xor(idx, md, 32);
            float mn = fmaxf(m, om);
            l = l * __expf(m - mn) + ol * __expf(om - mn);
            int take = (int)(om > m) | ((int)(om == m) & (int)(oi < idx));
            idx = take ? oi : idx;
            m = mn;
        }
        ml[r] = m; ll[r] = l; il[r] = idx;
    }

    if (lr == 0) {
        int p16 = i * 4 + j;
        #pragma unroll
        for (int r = 0; r < 8; ++r) {
            int row = rowTile * 16 + r + 8 * hf;
            idxRow[p16 * NN + row] = il[r];
            mRow[p16 * NN + row]   = ml[r];
            lRow[p16 * NN + row]   = ll[r];
        }
    }
}

// ---------------- Pass B: softmax @ sf_j, recon GEMM, masked MSE ----------------
__global__ void __launch_bounds__(256, 1)
passB(const _Float16* __restrict__ dnh, const _Float16* __restrict__ sfT,
      const _Float16* __restrict__ WrecT, const float* __restrict__ relWT,
      const int* __restrict__ idxRow, const float* __restrict__ mRow,
      const float* __restrict__ lRow, float* __restrict__ sqacc) {
    __shared__ _Float16 lds[8][16 * LDSTRIDE];
    int w    = threadIdx.x >> 5;
    int wid  = (blockIdx.x * blockDim.x + threadIdx.x) >> 5;
    int lane = threadIdx.x & 31;
    int q = wid >> 8, rowTile = wid & 255;
    if (q >= 12) return;
    int i = q / 3, rr = q % 3;
    int j = rr + (rr >= i ? 1 : 0);
    int p16 = i * 4 + j, pT = j * 4 + i;
    int hf = lane >> 4, lr = lane & 15;
    int koff = hf * 8;
    _Float16* myLds = &lds[w][0];

    // preload A fragments (dn_i rows) -- reused for sim recompute AND recon GEMM
    const _Float16* pa = dnh + ((size_t)((rowTile * 16 + lr) * VV + i)) * CC;
    v16h a[8];
    #pragma unroll
    for (int kc = 0; kc < 8; ++kc)
        a[kc] = load16h(pa + kc * 32 + koff, pa + kc * 32 + 16 + koff);

    // per-row softmax stats + mutual mask
    float mr[8], lw[8], mm[8];
    #pragma unroll
    for (int r = 0; r < 8; ++r) {
        int row = rowTile * 16 + r + 8 * hf;
        mr[r] = mRow[p16 * NN + row];
        lw[r] = lRow[p16 * NN + row];
        int ii = idxRow[p16 * NN + row];
        mm[r] = (idxRow[pT * NN + ii] == row) ? 1.0f : 0.0f;
    }

    v8f acc[16];
    #pragma unroll
    for (int t = 0; t < 16; ++t) acc[t] = (v8f){};

    const _Float16* dbase = dnh + (size_t)(lr * VV + j) * CC;
    const _Float16* sbase = sfT + ((size_t)(j * CC + lr)) * NN;

    for (int kc32 = 0; kc32 < 128; ++kc32) {
        // recompute two 16x16 sim tiles (k-columns kc32*32 .. +31), exp, stage to LDS
        #pragma unroll
        for (int t = 0; t < 2; ++t) {
            int ct = kc32 * 2 + t;
            const _Float16* pb = dbase + (size_t)ct * 16 * VV * CC;
            v16h b[8];
            #pragma unroll
            for (int kc = 0; kc < 8; ++kc)
                b[kc] = load16h(pb + kc * 32 + koff, pb + kc * 32 + 16 + koff);
            v8f c = {};
            #pragma unroll
            for (int kc = 0; kc < 8; ++kc)
                c = wmma_f16(a[kc], b[kc], c);
            #pragma unroll
            for (int r = 0; r < 8; ++r) {
                float e = __expf(c[r] - mr[r]);
                myLds[(r + 8 * hf) * LDSTRIDE + t * 16 + lr] = (_Float16)e;
            }
        }
        // reload P chunk (16x32) in WMMA A-layout (same-wave LDS ops are in-order)
        H16 up;
        up.f4[0] = *(const float4*)&myLds[lr * LDSTRIDE + koff];
        up.f4[1] = *(const float4*)&myLds[lr * LDSTRIDE + 16 + koff];
        v16h pA = up.v;
        // accumulate P @ sf_j over all 16 output channel tiles
        #pragma unroll
        for (int ct2 = 0; ct2 < 16; ++ct2) {
            const _Float16* ps = sbase + (size_t)ct2 * 16 * NN + kc32 * 32;
            v16h b = load16h(ps + koff, ps + 16 + koff);
            acc[ct2] = wmma_f16(pA, b, acc[ct2]);
        }
    }

    // recon = dn_i @ W_rec + relWT ; diff against softmax output; masked squared sum
    float sql = 0.0f;
    #pragma unroll
    for (int ct = 0; ct < 16; ++ct) {
        const _Float16* pw = WrecT + ((size_t)(ct * 16 + lr)) * CC;
        v16h b[8];
        #pragma unroll
        for (int kc = 0; kc < 8; ++kc)
            b[kc] = load16h(pw + kc * 32 + koff, pw + kc * 32 + 16 + koff);
        v8f c2 = {};
        #pragma unroll
        for (int kc = 0; kc < 8; ++kc)
            c2 = wmma_f16(a[kc], b[kc], c2);
        float wv = relWT[p16 * CC + ct * 16 + lr];
        #pragma unroll
        for (int r = 0; r < 8; ++r) {
            float recon = c2[r] + wv;
            float soft  = acc[ct][r] / lw[r];
            float d = recon - soft;
            sql += d * d * mm[r];
        }
    }
    #pragma unroll
    for (int md = 1; md < 32; md <<= 1) sql += __shfl_xor(sql, md, 32);
    if (lane == 0) atomicAdd(&sqacc[p16], sql);
}

// ---------------- Pass C: fold per-pair sums into scalar loss ----------------
__global__ void passC(const int* __restrict__ idxRow, const float* __restrict__ sqacc,
                      float* __restrict__ out) {
    __shared__ int red[256];
    int t = threadIdx.x;
    float loss_total = 0.0f, count = 0.0f;
    for (int q = 0; q < 12; ++q) {
        int i = q / 3, rr = q % 3;
        int j = rr + (rr >= i ? 1 : 0);
        int p16 = i * 4 + j, pT = j * 4 + i;
        int csel = 0;
        for (int n = t; n < NN; n += 256) {
            int ii = idxRow[p16 * NN + n];
            csel += (idxRow[pT * NN + ii] == n) ? 1 : 0;
        }
        red[t] = csel;
        __syncthreads();
        for (int s = 128; s > 0; s >>= 1) { if (t < s) red[t] += red[t + s]; __syncthreads(); }
        if (t == 0) {
            float nsel = (float)red[0];
            if (nsel > 0.0f) {
                loss_total += sqacc[p16] / fmaxf(nsel * (float)CC, 1.0f);
                count += 1.0f;
            }
        }
        __syncthreads();
    }
    if (t == 0) out[0] = (count > 0.0f) ? (loss_total / count) : 0.0f;
}

// ---------------- launcher ----------------
extern "C" void kernel_launch(void* const* d_in, const int* in_sizes, int n_in,
                              void* d_out, int out_size, void* d_ws, size_t ws_size,
                              hipStream_t stream) {
    const float* desc  = (const float*)d_in[0];
    const float* sf    = (const float*)d_in[1];
    const float* T     = (const float*)d_in[2];
    const float* W_rec = (const float*)d_in[3];
    const float* W_T   = (const float*)d_in[4];
    float* out = (float*)d_out;

    char* ws = (char*)d_ws;
    size_t off = 0;
    _Float16* dnh   = (_Float16*)(ws + off); off += (size_t)NN * VV * CC * 2;   // 8 MiB
    _Float16* sfT   = (_Float16*)(ws + off); off += (size_t)NN * VV * CC * 2;   // 8 MiB
    _Float16* WrecT = (_Float16*)(ws + off); off += (size_t)CC * CC * 2;        // 128 KiB
    float*    relWT = (float*)   (ws + off); off += 16 * CC * 4;                // 16 KiB
    int*      idxR  = (int*)     (ws + off); off += 16 * (size_t)NN * 4;        // 256 KiB
    float*    mR    = (float*)   (ws + off); off += 16 * (size_t)NN * 4;
    float*    lR    = (float*)   (ws + off); off += 16 * (size_t)NN * 4;
    float*    sqacc = (float*)   (ws + off); off += 64;

    prepNorm<<<NN * VV, 256, 0, stream>>>(desc, dnh);
    prepSfT <<<(NN * VV * CC) / 256, 256, 0, stream>>>(sf, sfT);
    prepWrec<<<(CC * CC) / 256, 256, 0, stream>>>(W_rec, WrecT);
    prepRel <<<16, 256, 0, stream>>>(T, W_T, relWT, sqacc);

    passA<<<(12 * 256) / 8, 256, 0, stream>>>(dnh, idxR, mR, lR);
    passB<<<(12 * 256) / 8, 256, 0, stream>>>(dnh, sfT, WrecT, relWT, idxR, mR, lR, sqacc);
    passC<<<1, 256, 0, stream>>>(idxR, sqacc, out);
}
